// KANMAMMOTE_71236327571717
// MI455X (gfx1250) — matrix operand order repaired and checked by reading, not compiled, not run on hardware
//
#include <hip/hip_runtime.h>
#include <hip/hip_bf16.h>

typedef __attribute__((ext_vector_type(16))) _Float16 v16h;
typedef __attribute__((ext_vector_type(8)))  float    v8f;

#define Bz   4
#define Sz   512
#define FIN  256
#define Dz   1024
#define DINz 2048
#define Nz   8
#define Ez   8
#define F2z  32
#define Lz   3
#define ODz  256
#define BS   (Bz*Sz)          // 2048 tokens

// ---------------------------------------------------------------------------
// WMMA GEMM: C[M,N] = A[M,K] * B[K,N] (+bias[N]) (+resid[M,N])
// f32 storage, f16 WMMA inputs, f32 accumulate. M%64==0, N%(16*NT)==0, K%64==0.
// Block = 128 threads = 4 waves. Block tile: 64 rows (one 16-row tile/wave)
// x (16*NT) cols (shared). A 64-deep B K-block (64 x 16*NT) is staged in LDS
// as f16, column-major, so each lane's WMMA B fragment is one contiguous
// 32-byte ds read. Per stage: 8 coalesced global_load_b128 issued up front,
// then 2 K-sub-blocks x NT wmma ops, each A fragment reused NT times.
// ---------------------------------------------------------------------------
template <int NT, bool HAS_BIAS, bool HAS_RESID>
__global__ __launch_bounds__(128)
void gemm_wmma_f16(const float* __restrict__ A, const float* __restrict__ Bm,
                   const float* __restrict__ bias, const float* __restrict__ resid,
                   float* __restrict__ C, int M, int N, int K)
{
  __shared__ _Float16 Bs[64 * 64];              // [col][k], max NT=4 -> 8 KB

  const int tid    = threadIdx.x;
  const int lane   = tid & 31;
  const int wave   = tid >> 5;
  const int nblk   = blockIdx.x * (16 * NT);    // block's first N column
  const int m0     = (blockIdx.y * 4 + wave) * 16;   // exact launch: m0 < M

  const int half16 = lane >> 4;       // 0: lanes 0-15, 1: lanes 16-31
  const int l15    = lane & 15;
  const int am     = m0 + l15;        // A-matrix row held by this lane
  const int khA    = half16 * 16;     // A k-base (contiguous k = 16h + s, matches B)
  const int kbB    = half16 * 16;     // B k-base (contiguous k = 16h + s, matches A)

  v8f acc[NT];
#pragma unroll
  for (int t = 0; t < NT; ++t) acc[t] = (v8f){};

  const float* Abase = A + (long long)am * K + khA;

  for (int kk = 0; kk < K; kk += 64) {
    // ---- issue all stage loads first (2*NT x global_load_b128 / thread) ---
    float4 vv[2 * NT];
#pragma unroll
    for (int t = 0; t < 2 * NT; ++t) {
      const int q  = tid + t * 128;             // 0 .. 256*NT-1
      const int k  = q / (4 * NT);              // 0..63
      const int c4 = q % (4 * NT);              // float4 column group
      vv[t] = *(const float4*)(Bm + (long long)(kk + k) * N + nblk + c4 * 4);
    }
    __syncthreads();                            // previous stage fully consumed
    // ---- convert + scatter into LDS, column-major f16 ---------------------
#pragma unroll
    for (int t = 0; t < 2 * NT; ++t) {
      const int q  = tid + t * 128;
      const int k  = q / (4 * NT);
      const int cb = (q % (4 * NT)) * 4;
      Bs[(cb + 0) * 64 + k] = (_Float16)vv[t].x;
      Bs[(cb + 1) * 64 + k] = (_Float16)vv[t].y;
      Bs[(cb + 2) * 64 + k] = (_Float16)vv[t].z;
      Bs[(cb + 3) * 64 + k] = (_Float16)vv[t].w;
    }
    __syncthreads();

    // ---- 2 K-sub-blocks x NT wmma -----------------------------------------
#pragma unroll
    for (int ks = 0; ks < 2; ++ks) {
      const int kof = kk + ks * 32;
      const float4 a0 = *(const float4*)(Abase + kof);        // K = khA+0..3
      const float4 a1 = *(const float4*)(Abase + kof + 4);    // K = khA+4..7
      const float4 a2 = *(const float4*)(Abase + kof + 8);    // K = khA+8..11
      const float4 a3 = *(const float4*)(Abase + kof + 12);   // K = khA+12..15
      v16h a;
      a[0]  = (_Float16)a0.x; a[1]  = (_Float16)a0.y;
      a[2]  = (_Float16)a0.z; a[3]  = (_Float16)a0.w;
      a[4]  = (_Float16)a1.x; a[5]  = (_Float16)a1.y;
      a[6]  = (_Float16)a1.z; a[7]  = (_Float16)a1.w;
      a[8]  = (_Float16)a2.x; a[9]  = (_Float16)a2.y;
      a[10] = (_Float16)a2.z; a[11] = (_Float16)a2.w;
      a[12] = (_Float16)a3.x; a[13] = (_Float16)a3.y;
      a[14] = (_Float16)a3.z; a[15] = (_Float16)a3.w;
#pragma unroll
      for (int t = 0; t < NT; ++t) {
        const int col = t * 16 + l15;
        const v16h b = *(const v16h*)&Bs[col * 64 + ks * 32 + kbB];  // 32B
        acc[t] = __builtin_amdgcn_wmma_f32_16x16x32_f16(false, a, false, b,
                                                        (short)0, acc[t], false, false);
      }
    }
  }

  // ---- epilogue (branch-free per instantiation) ---------------------------
#pragma unroll
  for (int t = 0; t < NT; ++t) {
    const int bn = nblk + t * 16 + l15;
    float bv = 0.0f;
    if (HAS_BIAS) bv = bias[bn];
#pragma unroll
    for (int r = 0; r < 8; ++r) {
      const int row = m0 + r + half16 * 8;     // ISA C/D layout
      const long long o = (long long)row * N + bn;
      float v = acc[t][r] + bv;
      if (HAS_RESID) v += resid[o];
      C[o] = v;
    }
  }
}

// ---------------------------------------------------------------------------
__global__ void dt_kernel(const float* __restrict__ ts, float* __restrict__ dt)
{
  int t = blockIdx.x * blockDim.x + threadIdx.x;
  if (t >= BS) return;
  int s = t & (Sz - 1);
  dt[t] = (s == 0) ? 0.0f : (ts[t] - ts[t - 1]);
}

__global__ void zero_kernel(float* __restrict__ p, int n)
{
  for (int i = blockIdx.x * blockDim.x + threadIdx.x; i < n; i += gridDim.x * blockDim.x)
    p[i] = 0.0f;
}

// Pack Wr(D,E) | W_B(D,N) | W_C(D,N) | zeros  into a (D x 32) matrix.
__global__ void pack_wrbc_kernel(const float* __restrict__ Wr, const float* __restrict__ WB,
                                 const float* __restrict__ WC, float* __restrict__ out)
{
  int idx = blockIdx.x * blockDim.x + threadIdx.x;  // D*32
  if (idx >= Dz * 32) return;
  int r = idx >> 5, c = idx & 31;
  float v = 0.0f;
  if (c < 8)       v = Wr[r * Ez + c];
  else if (c < 16) v = WB[r * Nz + (c - 8)];
  else if (c < 24) v = WC[r * Nz + (c - 16)];
  out[idx] = v;
}

// Router: top-2 of logits, softmax over the 2, emit dense gates/mask (layer 0),
// and fill the expanded sparse Fourier basis A_exp[t, e*64 + f] = g_e * {sin,cos}.
__global__ void router_kernel(const float* __restrict__ lbc, const float* __restrict__ dtv,
                              const float* __restrict__ freqs, float* __restrict__ Aexp,
                              float* __restrict__ gates0, float* __restrict__ mask0,
                              int save)
{
  int t = blockIdx.x * blockDim.x + threadIdx.x;
  if (t >= BS) return;
  float lg[Ez];
#pragma unroll
  for (int e = 0; e < Ez; ++e) lg[e] = lbc[t * 32 + e];
  int i1 = 0; float v1 = lg[0];
#pragma unroll
  for (int e = 1; e < Ez; ++e) if (lg[e] > v1) { v1 = lg[e]; i1 = e; }
  int i2 = (i1 == 0) ? 1 : 0; float v2 = lg[i2];
#pragma unroll
  for (int e = 0; e < Ez; ++e) if (e != i1 && lg[e] > v2) { v2 = lg[e]; i2 = e; }
  float g1 = 1.0f / (1.0f + expf(v2 - v1));   // softmax over {v1,v2}, v1 >= v2
  float g2 = 1.0f - g1;
  if (save) {
#pragma unroll
    for (int e = 0; e < Ez; ++e) {
      gates0[t * Ez + e] = (e == i1) ? g1 : ((e == i2) ? g2 : 0.0f);
      mask0 [t * Ez + e] = (e == i1 || e == i2) ? 1.0f : 0.0f;
    }
  }
  float d = dtv[t];
  float* row = Aexp + (long long)t * (Ez * 2 * F2z);
#pragma unroll 4
  for (int f = 0; f < F2z; ++f) {
    float ang = d * freqs[f];
    float sn = sinf(ang), cs = cosf(ang);
    row[i1 * 64 + f]        = g1 * sn;
    row[i1 * 64 + F2z + f]  = g1 * cs;
    row[i2 * 64 + f]        = g2 * sn;
    row[i2 * 64 + F2z + f]  = g2 * cs;
  }
}

__device__ __forceinline__ float softplusf(float v)
{ return (v > 20.0f) ? v : log1pf(expf(v)); }

// delta = softplus(x * a_d + (te@W_dt) + b_d); dl holds te@W_dt on entry.
__global__ void delta_kernel(const float* __restrict__ x, float* __restrict__ dl,
                             const float* __restrict__ a_d, const float* __restrict__ b_d)
{
  const int total = BS * DINz;
  for (int idx = blockIdx.x * blockDim.x + threadIdx.x; idx < total;
       idx += gridDim.x * blockDim.x) {
    int i = idx & (DINz - 1);
    dl[idx] = softplusf(x[idx] * a_d[i] + dl[idx] + b_d[i]);
  }
}

// Selective-SSM scan: one thread per (batch, channel i). N=8 state in registers.
// Emits yz = y * silu(z) ready for the W_out GEMM.
__global__ __launch_bounds__(256)
void scan_kernel(const float* __restrict__ dl, const float* __restrict__ x,
                 const float* __restrict__ z, const float* __restrict__ lbc,
                 const float* __restrict__ A_log, float* __restrict__ yz)
{
  int id = blockIdx.x * blockDim.x + threadIdx.x;   // Bz*DINz threads
  if (id >= Bz * DINz) return;
  int b = id / DINz, i = id % DINz;
  float Adec[Nz];
#pragma unroll
  for (int n = 0; n < Nz; ++n) Adec[n] = softplusf(A_log[i * Nz + n]);
  float h[Nz];
#pragma unroll
  for (int n = 0; n < Nz; ++n) h[n] = 0.0f;

  for (int s = 0; s < Sz; ++s) {
    long long tok = (long long)b * Sz + s;
    long long o   = tok * DINz + i;
    float d  = dl[o];
    float dx = d * x[o];
    float zv = z[o];
    const float* bc = lbc + tok * 32;
    float y = 0.0f;
#pragma unroll
    for (int n = 0; n < Nz; ++n) {
      h[n] = expf(-Adec[n] * d) * h[n] + dx * bc[8 + n];
      y   += h[n] * bc[16 + n];
    }
    yz[o] = y * (zv / (1.0f + expf(-zv)));
  }
}

// Load-balance loss: single block reduces gates0/mask0 over 2048 tokens.
__global__ __launch_bounds__(256)
void lb_loss_kernel(const float* __restrict__ gates0, const float* __restrict__ mask0,
                    float* __restrict__ out3)
{
  __shared__ float sh[256];
  int tid = threadIdx.x;
  float sg[Ez], sm[Ez];
#pragma unroll
  for (int e = 0; e < Ez; ++e) { sg[e] = 0.0f; sm[e] = 0.0f; }
  for (int t = tid; t < BS; t += 256)
#pragma unroll
    for (int e = 0; e < Ez; ++e) { sg[e] += gates0[t * Ez + e]; sm[e] += mask0[t * Ez + e]; }

  float lb = 0.0f;
  for (int e = 0; e < Ez; ++e) {
    sh[tid] = sg[e]; __syncthreads();
    for (int o = 128; o > 0; o >>= 1) { if (tid < o) sh[tid] += sh[tid + o]; __syncthreads(); }
    float G = sh[0]; __syncthreads();
    sh[tid] = sm[e]; __syncthreads();
    for (int o = 128; o > 0; o >>= 1) { if (tid < o) sh[tid] += sh[tid + o]; __syncthreads(); }
    float Fm = sh[0]; __syncthreads();
    if (tid == 0) lb += (Fm / (float)BS) * (G / (float)BS);
  }
  if (tid == 0) { out3[0] = (float)Ez * lb; out3[1] = 0.0f; out3[2] = 0.0f; }
}

// Sobolev-L2 + total-variation over We (L,E,2F2,D); diff along the 2F2 axis.
__global__ __launch_bounds__(256)
void we_loss_kernel(const float* __restrict__ We, float* __restrict__ out3)
{
  __shared__ float s2sh[256], tvsh[256];
  int tid = threadIdx.x;
  const long long total = (long long)Lz * Ez * 2 * F2z * Dz;   // 1,572,864
  float s2 = 0.0f, tv = 0.0f;
  for (long long idx = (long long)blockIdx.x * 256 + tid; idx < total;
       idx += (long long)gridDim.x * 256) {
    int f = (int)((idx >> 10) & 63);          // D = 1024
    float w = We[idx];
    s2 += w * w;
    if (f < 63) tv += fabsf(We[idx + Dz] - w);
  }
  s2sh[tid] = s2; tvsh[tid] = tv; __syncthreads();
  for (int o = 128; o > 0; o >>= 1) {
    if (tid < o) { s2sh[tid] += s2sh[tid + o]; tvsh[tid] += tvsh[tid + o]; }
    __syncthreads();
  }
  if (tid == 0) {
    atomicAdd(&out3[1], 1e-4f * s2sh[0]);
    atomicAdd(&out3[2], 1e-4f * tvsh[0]);
  }
}

// ---------------------------------------------------------------------------
extern "C" void kernel_launch(void* const* d_in, const int* in_sizes, int n_in,
                              void* d_out, int out_size, void* d_ws, size_t ws_size,
                              hipStream_t stream)
{
  (void)in_sizes; (void)n_in; (void)out_size; (void)ws_size;
  const float* timestamps = (const float*)d_in[0];
  const float* features   = (const float*)d_in[1];
  const float* W_proj     = (const float*)d_in[2];
  const float* b_proj     = (const float*)d_in[3];
  const float* Wr         = (const float*)d_in[4];
  const float* freqs      = (const float*)d_in[5];
  const float* We         = (const float*)d_in[6];
  const float* W_in       = (const float*)d_in[7];
  const float* W_z        = (const float*)d_in[8];
  const float* a_d        = (const float*)d_in[9];
  const float* W_dt       = (const float*)d_in[10];
  const float* b_d        = (const float*)d_in[11];
  const float* A_log      = (const float*)d_in[12];
  const float* W_B        = (const float*)d_in[13];
  const float* W_C        = (const float*)d_in[14];
  const float* W_out      = (const float*)d_in[15];
  const float* W_head     = (const float*)d_in[16];
  const float* b_head     = (const float*)d_in[17];
  float* out = (float*)d_out;

  // workspace layout (f32)
  float* ws    = (float*)d_ws;
  float* dt    = ws;                         // 2048
  float* u     = dt    + BS;                 // 2048*1024
  float* lbc   = u     + BS * Dz;            // 2048*32  (logits|Bt|Ct|pad)
  float* g0    = lbc   + BS * 32;            // 2048*8
  float* m0    = g0    + BS * Ez;            // 2048*8
  float* Aexp  = m0    + BS * Ez;            // 2048*512
  float* te    = Aexp  + BS * (Ez * 2 * F2z);// 2048*1024
  float* x     = te    + BS * Dz;            // 2048*2048
  float* z     = x     + BS * DINz;          // 2048*2048
  float* dl    = z     + BS * DINz;          // 2048*2048
  float* yz    = dl    + BS * DINz;          // 2048*2048
  float* wrbc  = yz    + BS * DINz;          // 1024*32

  const int GB = 256;   // generic elementwise block

  // dt
  dt_kernel<<<(BS + 255) / 256, 256, 0, stream>>>(timestamps, dt);

  // u = features @ W_proj + b_proj   (2048 x 256 x 1024)
  gemm_wmma_f16<4, true, false><<<dim3(Dz / 64, BS / 64), 128, 0, stream>>>(
      features, W_proj, b_proj, nullptr, u, BS, Dz, FIN);

  for (int l = 0; l < Lz; ++l) {
    const float* Wr_l   = Wr    + (long long)l * Dz * Ez;
    const float* WB_l   = W_B   + (long long)l * Dz * Nz;
    const float* WC_l   = W_C   + (long long)l * Dz * Nz;
    const float* fr_l   = freqs + (long long)l * F2z;
    const float* We_l   = We    + (long long)l * Ez * 2 * F2z * Dz;
    const float* Win_l  = W_in  + (long long)l * Dz * DINz;
    const float* Wz_l   = W_z   + (long long)l * Dz * DINz;
    const float* Wdt_l  = W_dt  + (long long)l * Dz * DINz;
    const float* ad_l   = a_d   + (long long)l * DINz;
    const float* bd_l   = b_d   + (long long)l * DINz;
    const float* Alog_l = A_log + (long long)l * DINz * Nz;
    const float* Wout_l = W_out + (long long)l * DINz * Dz;

    // pack router + B + C projections into one 1024x32 matrix
    pack_wrbc_kernel<<<(Dz * 32 + GB - 1) / GB, GB, 0, stream>>>(Wr_l, WB_l, WC_l, wrbc);
    // lbc = u @ wrbc   (2048 x 1024 x 32), NT=2 -> one block column
    gemm_wmma_f16<2, false, false><<<dim3(1, BS / 64), 128, 0, stream>>>(
        u, wrbc, nullptr, nullptr, lbc, BS, 32, Dz);

    // expanded sparse Fourier basis
    zero_kernel<<<512, GB, 0, stream>>>(Aexp, BS * Ez * 2 * F2z);
    router_kernel<<<(BS + GB - 1) / GB, GB, 0, stream>>>(
        lbc, dt, fr_l, Aexp, g0, m0, (l == 0) ? 1 : 0);

    // te = Aexp @ We_flat   (2048 x 512 x 1024)
    gemm_wmma_f16<4, false, false><<<dim3(Dz / 64, BS / 64), 128, 0, stream>>>(
        Aexp, We_l, nullptr, nullptr, te, BS, Dz, Ez * 2 * F2z);

    // x = u @ W_in ; z = u @ W_z ; dl = te @ W_dt     (2048 x 1024 x 2048)
    gemm_wmma_f16<4, false, false><<<dim3(DINz / 64, BS / 64), 128, 0, stream>>>(
        u, Win_l, nullptr, nullptr, x, BS, DINz, Dz);
    gemm_wmma_f16<4, false, false><<<dim3(DINz / 64, BS / 64), 128, 0, stream>>>(
        u, Wz_l, nullptr, nullptr, z, BS, DINz, Dz);
    gemm_wmma_f16<4, false, false><<<dim3(DINz / 64, BS / 64), 128, 0, stream>>>(
        te, Wdt_l, nullptr, nullptr, dl, BS, DINz, Dz);

    // delta = softplus(x*a_d + dl + b_d)
    delta_kernel<<<1024, GB, 0, stream>>>(x, dl, ad_l, bd_l);

    // SSM scan -> yz = y * silu(z)
    scan_kernel<<<(Bz * DINz + 255) / 256, 256, 0, stream>>>(dl, x, z, lbc, Alog_l, yz);

    // u = yz @ W_out + u   (2048 x 2048 x 1024, residual, in-place safe)
    gemm_wmma_f16<4, false, true><<<dim3(Dz / 64, BS / 64), 128, 0, stream>>>(
        yz, Wout_l, nullptr, u, u, BS, Dz, DINz);
  }

  // head: out = u @ W_head + b_head  (2048 x 1024 x 256)
  gemm_wmma_f16<4, true, false><<<dim3(ODz / 64, BS / 64), 128, 0, stream>>>(
      u, W_head, b_head, nullptr, out, BS, ODz, Dz);

  // losses -> out[BS*OD .. +2]
  float* out3 = out + (long long)BS * ODz;
  lb_loss_kernel<<<1, 256, 0, stream>>>(g0, m0, out3);
  we_loss_kernel<<<256, 256, 0, stream>>>(We, out3);
}
